// SurfSageEncoder_40999757808027
// MI455X (gfx1250) — matrix operand order repaired and checked by reading, not compiled, not run on hardware
//
#include <hip/hip_runtime.h>
#include <hip/hip_bf16.h>
#include <stdint.h>

// ---------------- WMMA vector types (gfx1250, wave32) ----------------
typedef __attribute__((ext_vector_type(16))) __bf16 v16bf;
typedef __attribute__((ext_vector_type(8)))  float  v8f;

#define N_NODES 50000
#define N_EDGES 600000

// Use gfx1250 async global->LDS copy (inline asm, ISA §15.18 op 98). Set to 0 to
// fall back to a plain cooperative LDS copy if the assembler rejects it.
#define USE_ASYNC_LDS 1

static __device__ __forceinline__ unsigned short f32_to_bf16(float f) {
    unsigned int u = __float_as_uint(f);
    u += 0x7fffu + ((u >> 16) & 1u);            // round-to-nearest-even
    return (unsigned short)(u >> 16);
}
static __device__ __forceinline__ unsigned int pack2bf(float a, float b) {
    return (unsigned int)f32_to_bf16(a) | ((unsigned int)f32_to_bf16(b) << 16);
}

// ---------------- utility kernels ----------------
__global__ void fill_f32(float* __restrict__ p, float v, int n4) {
    int i = blockIdx.x * blockDim.x + threadIdx.x;
    if (i < n4) ((float4*)p)[i] = make_float4(v, v, v, v);
}

__global__ void degree_kernel(const long long* __restrict__ dst, float* __restrict__ deg, int E) {
    int e = blockIdx.x * blockDim.x + threadIdx.x;
    if (e < E) {
        int d = (int)dst[e];
        __hip_atomic_fetch_add(&deg[d], 1.0f, __ATOMIC_RELAXED, __HIP_MEMORY_SCOPE_AGENT);
    }
}

// one wave per edge; each lane handles 4 floats of the 128-wide row (coalesced 16B loads)
__global__ void scatter_add_kernel(const float* __restrict__ X,
                                   const long long* __restrict__ src,
                                   const long long* __restrict__ dst,
                                   float* __restrict__ agg, int E) {
    int idx = blockIdx.x * blockDim.x + threadIdx.x;
    if (idx >= E * 32) return;
    int e = idx >> 5;
    int c = (idx & 31) << 2;
    int s = (int)src[e];
    int d = (int)dst[e];
    float4 v = *(const float4*)(X + (size_t)s * 128 + c);
    float* base = agg + (size_t)d * 128 + c;
    __hip_atomic_fetch_add(base + 0, v.x, __ATOMIC_RELAXED, __HIP_MEMORY_SCOPE_AGENT);
    __hip_atomic_fetch_add(base + 1, v.y, __ATOMIC_RELAXED, __HIP_MEMORY_SCOPE_AGENT);
    __hip_atomic_fetch_add(base + 2, v.z, __ATOMIC_RELAXED, __HIP_MEMORY_SCOPE_AGENT);
    __hip_atomic_fetch_add(base + 3, v.w, __ATOMIC_RELAXED, __HIP_MEMORY_SCOPE_AGENT);
}

// A[i, 0:K) = bf16(agg[i]*inv_deg) ; A[i, K:2K) = bf16(Xin[i])   (K == 128 here)
__global__ void build_A_kernel(const float* __restrict__ agg, const float* __restrict__ Xin,
                               const float* __restrict__ deg,
                               unsigned short* __restrict__ A, int Nn, int K) {
    int per = K >> 2;
    int idx = blockIdx.x * blockDim.x + threadIdx.x;
    if (idx >= Nn * per) return;
    int i = idx / per;
    int c = (idx % per) << 2;
    float inv = 1.0f / fmaxf(deg[i], 1.0f);
    float4 m = *(const float4*)(agg + (size_t)i * K + c);
    float4 x = *(const float4*)(Xin + (size_t)i * K + c);
    uint2 pm; pm.x = pack2bf(m.x * inv, m.y * inv); pm.y = pack2bf(m.z * inv, m.w * inv);
    uint2 px; px.x = pack2bf(x.x, x.y);             px.y = pack2bf(x.z, x.w);
    unsigned short* row = A + (size_t)i * 2 * K;
    *(uint2*)(row + c)     = pm;
    *(uint2*)(row + K + c) = px;
}

// Wcat[r, 0:K) = bf16(Wl[r,:]) ; Wcat[r, K:2K) = bf16(Wr[r,:])
__global__ void pack_w2_kernel(const float* __restrict__ Wl, const float* __restrict__ Wr,
                               unsigned short* __restrict__ out, int Dout, int K) {
    int idx = blockIdx.x * blockDim.x + threadIdx.x;
    if (idx >= Dout * K) return;
    int r = idx / K, c = idx % K;
    out[(size_t)r * 2 * K + c]     = f32_to_bf16(Wl[idx]);
    out[(size_t)r * 2 * K + K + c] = f32_to_bf16(Wr[idx]);
}

// plain elementwise f32 -> bf16 (used for h3 activations and Wreg)
__global__ void cvt_bf16_kernel(const float* __restrict__ in, unsigned short* __restrict__ out, int n4) {
    int i = blockIdx.x * blockDim.x + threadIdx.x;
    if (i >= n4) return;
    float4 v = ((const float4*)in)[i];
    uint2 p; p.x = pack2bf(v.x, v.y); p.y = pack2bf(v.z, v.w);
    ((uint2*)out)[i] = p;
}

// ---------------- WMMA GEMM: out[M,Nout] = A[M,KK] @ W[Nout,KK]^T + bias (+ReLU) ----------------
// W (<= 64KB of bf16) is staged into LDS once per block via async global->LDS copy,
// then read as ds_load_b128 fragments. A global loads are software-pipelined (kb+1
// prefetched during the WMMAs of kb). One wave per 16-row M tile; A fragment reused
// across NT = Nout/16 column tiles.
// A operand layout (16-bit, 16x32): lane-half 0 holds K[0..7],[16..23]; half 1 holds K[8..15],[24..31].
// B operand layout (16-bit, 32x16): lane n = output col n = row n of W; same K split as A.
template <int NT, int KK>
__global__ void sage_gemm_wmma(const unsigned short* __restrict__ A,
                               const unsigned short* __restrict__ W,
                               const float* __restrict__ bias,
                               float* __restrict__ out,
                               int Mtiles, int relu) {
    constexpr int Nout = NT * 16;
    constexpr int KB   = KK / 32;
    __shared__ unsigned short sB[Nout * KK];

    // ---- stage W[Nout,KK] into LDS ----
    {
        const int chunks = (Nout * KK) / 8;          // 16B chunks
#if USE_ASYNC_LDS
        for (int i = threadIdx.x; i < chunks; i += blockDim.x) {
            unsigned long long g = (unsigned long long)(uintptr_t)(W + (size_t)i * 8);
            unsigned int       l = (unsigned int)(uintptr_t)(&sB[(size_t)i * 8]); // addr[31:0] = LDS offset
            asm volatile("global_load_async_to_lds_b128 %0, %1, off"
                         :: "v"(l), "v"(g) : "memory");
        }
        asm volatile("s_wait_asynccnt 0x0" ::: "memory");
#else
        for (int i = threadIdx.x; i < chunks; i += blockDim.x)
            ((uint4*)sB)[i] = ((const uint4*)W)[i];
#endif
        __syncthreads();
    }

    const int lane = threadIdx.x & 31;
    const int wv   = threadIdx.x >> 5;
    int mtile = blockIdx.x * (blockDim.x >> 5) + wv;
    if (mtile >= Mtiles) mtile = Mtiles - 1;   // clamp (redundant identical work) keeps EXEC all-ones
    const int half = lane >> 4;
    const int l    = lane & 15;

    v8f zero = {};
    v8f acc[NT];
#pragma unroll
    for (int t = 0; t < NT; ++t) acc[t] = zero;

    const unsigned short* arow = A + (size_t)(mtile * 16 + l) * KK + half * 8;

    // software pipeline on A (global); B fragments come from LDS
    uint4 ca0 = ((const uint4*)arow)[0];
    uint4 ca1 = ((const uint4*)arow)[2];               // +16 elements = +32B
#pragma unroll
    for (int kb = 0; kb < KB; ++kb) {
        uint4 na0, na1;
        if (kb + 1 < KB) {
            const uint4* np = (const uint4*)(arow + (kb + 1) * 32);
            na0 = np[0];
            na1 = np[2];
        }
        union { uint4 u[2]; v16bf v; } a;
        a.u[0] = ca0;                                  // K = kbase .. kbase+7
        a.u[1] = ca1;                                  // K = kbase+16 .. kbase+23
#pragma unroll
        for (int t = 0; t < NT; ++t) {
            union { uint4 u[2]; v16bf v; } b;
            const uint4* bp = (const uint4*)(sB + (size_t)(t * 16 + l) * KK + kb * 32 + half * 8);
            b.u[0] = bp[0];                            // K = kbase .. kbase+7, kbase+16 .. kbase+23
            b.u[1] = bp[2];
            acc[t] = __builtin_amdgcn_wmma_f32_16x16x32_bf16(
                false, a.v, false, b.v, (short)0, acc[t], false, false);
        }
        ca0 = na0;
        ca1 = na1;
    }

#pragma unroll
    for (int t = 0; t < NT; ++t) {
        int col = t * 16 + l;
        float bv = bias[col];
#pragma unroll
        for (int r = 0; r < 8; ++r) {
            int row = mtile * 16 + r + half * 8;       // C/D layout: VGPR r, lane-half selects M / M+8
            float v = acc[t][r] + bv;
            if (relu) v = fmaxf(v, 0.0f);
            out[(size_t)row * Nout + col] = v;
        }
    }
}

// ---------------- per-layer driver ----------------
static void run_sage_layer(const float* Xin, int K, int Dout,
                           const float* Wl, const float* bl, const float* Wr,
                           float* Hout, int relu,
                           const long long* src, const long long* dst,
                           float* deg, float* agg, unsigned short* Abuf, unsigned short* Wcat,
                           hipStream_t stream) {
    int aggN4 = N_NODES * K / 4;
    fill_f32<<<(aggN4 + 255) / 256, 256, 0, stream>>>(agg, 0.0f, aggN4);
    int sth = N_EDGES * 32;
    scatter_add_kernel<<<(sth + 255) / 256, 256, 0, stream>>>(Xin, src, dst, agg, N_EDGES);
    int bth = N_NODES * (K / 4);
    build_A_kernel<<<(bth + 255) / 256, 256, 0, stream>>>(agg, Xin, deg, Abuf, N_NODES, K);
    int wth = Dout * K;
    pack_w2_kernel<<<(wth + 255) / 256, 256, 0, stream>>>(Wl, Wr, Wcat, Dout, K);

    int Mtiles = N_NODES / 16;                 // 3125, exact
    int blocks = (Mtiles + 3) / 4;             // 4 waves / block
    if (Dout == 128)
        sage_gemm_wmma<8, 256><<<blocks, 128, 0, stream>>>(Abuf, Wcat, bl, Hout, Mtiles, relu);
    else
        sage_gemm_wmma<4, 256><<<blocks, 128, 0, stream>>>(Abuf, Wcat, bl, Hout, Mtiles, relu);
}

extern "C" void kernel_launch(void* const* d_in, const int* in_sizes, int n_in,
                              void* d_out, int out_size, void* d_ws, size_t ws_size,
                              hipStream_t stream) {
    (void)in_sizes; (void)n_in; (void)out_size; (void)ws_size;

    const float*     x    = (const float*)d_in[0];
    const long long* ei   = (const long long*)d_in[1];   // int64 [2,E]
    const long long* src  = ei;
    const long long* dst  = ei + N_EDGES;
    const float* Wl1 = (const float*)d_in[2];
    const float* bl1 = (const float*)d_in[3];
    const float* Wr1 = (const float*)d_in[4];
    const float* Wl2 = (const float*)d_in[5];
    const float* bl2 = (const float*)d_in[6];
    const float* Wr2 = (const float*)d_in[7];
    const float* Wl3 = (const float*)d_in[8];
    const float* bl3 = (const float*)d_in[9];
    const float* Wr3 = (const float*)d_in[10];
    const float* Wreg = (const float*)d_in[11];
    const float* breg = (const float*)d_in[12];
    float* out = (float*)d_out;

    // workspace carve-out
    char* ws = (char*)d_ws;
    size_t off = 0;
    auto carve = [&](size_t bytes) -> void* {
        void* p = ws + off;
        off += (bytes + 255) & ~(size_t)255;
        return p;
    };
    float*          deg  = (float*)carve((size_t)N_NODES * 4);
    float*          agg  = (float*)carve((size_t)N_NODES * 128 * 4);
    unsigned short* Abuf = (unsigned short*)carve((size_t)N_NODES * 256 * 2);
    float*          h1   = (float*)carve((size_t)N_NODES * 128 * 4);
    float*          h2   = (float*)carve((size_t)N_NODES * 128 * 4);
    unsigned short* Wcat = (unsigned short*)carve((size_t)128 * 256 * 2);
    float*          h3   = h1;   // layer-3 output [N,64] reuses h1's space

    // degrees (shared by all layers)
    fill_f32<<<(N_NODES / 4 + 255) / 256, 256, 0, stream>>>(deg, 0.0f, N_NODES / 4);
    degree_kernel<<<(N_EDGES + 255) / 256, 256, 0, stream>>>(dst, deg, N_EDGES);

    run_sage_layer(x,  128, 128, Wl1, bl1, Wr1, h1, 1, src, dst, deg, agg, Abuf, Wcat, stream);
    run_sage_layer(h1, 128, 128, Wl2, bl2, Wr2, h2, 1, src, dst, deg, agg, Abuf, Wcat, stream);
    run_sage_layer(h2, 128,  64, Wl3, bl3, Wr3, h3, 0, src, dst, deg, agg, Abuf, Wcat, stream);

    // regression head: out = h3 @ Wreg^T + breg   (K = 64)
    int n4a = N_NODES * 64 / 4;
    cvt_bf16_kernel<<<(n4a + 255) / 256, 256, 0, stream>>>(h3, Abuf, n4a);
    int n4w = 64 * 64 / 4;
    cvt_bf16_kernel<<<(n4w + 255) / 256, 256, 0, stream>>>(Wreg, Wcat, n4w);
    int Mtiles = N_NODES / 16;
    int blocks = (Mtiles + 3) / 4;
    sage_gemm_wmma<4, 64><<<blocks, 128, 0, stream>>>(Abuf, Wcat, breg, out, Mtiles, 0);
}
